// LSLGL_60189671686395
// MI455X (gfx1250) — compile-verified, matrix operand order checked by reading, not run on hardware
//
#include <hip/hip_runtime.h>
#include <hip/hip_bf16.h>
#include <hip/hip_fp16.h>

typedef __attribute__((ext_vector_type(16))) _Float16 v16h;
typedef __attribute__((ext_vector_type(8)))  float    v8f;

#define BN_EPS 1e-5f

__device__ __forceinline__ float sgnroot(float a) {
  float r = fmaxf(sqrtf(fabsf(a)), 1e-8f);
  return a > 0.f ? r : (a < 0.f ? -r : 0.f);   // sign(0)=0 like jnp.sign
}
__device__ __forceinline__ float softsign(float h) { return h / (1.f + fabsf(h)); }

// K-index interleave for 16-bit A/B fragments (ISA 7.12.2, 16x32 f16):
// lanes 0-15: K = e(0..7) then 16+e; lanes 16-31: shifted by +8.
__device__ __forceinline__ int kmap(int e, int half) {
  return (e & 7) + ((e & 8) << 1) + (half << 3);
}

// A fragment (16x32 f16) from row-major f16 LDS, leading dim lda
__device__ __forceinline__ v16h frag_a_lds(const _Float16* a, int lda, int m0, int k0, int lane) {
  int half = lane >> 4, m = m0 + (lane & 15);
  v16h f;
#pragma unroll
  for (int e = 0; e < 16; ++e) f[e] = a[m * lda + k0 + kmap(e, half)];
  return f;
}
// B fragment (32x16 f16) from row-major [K][N] f16 LDS, leading dim ldb
__device__ __forceinline__ v16h frag_b_lds(const _Float16* b, int ldb, int k0, int n0, int lane) {
  int half = lane >> 4, n = n0 + (lane & 15);
  v16h f;
#pragma unroll
  for (int e = 0; e < 16; ++e) f[e] = b[(k0 + kmap(e, half)) * ldb + n];
  return f;
}
// A fragment of the (never-materialized) adjacency:
// adj[m,y] = rinv[m] * sgnroot(x[m]*s[y] + s[m]*x[y]), all within channel block `base`
__device__ __forceinline__ v16h frag_a_adj(const float* xs, const float* ss, const float* rinv,
                                           int base, int m0, int k0, int lane) {
  int half = lane >> 4, m = m0 + (lane & 15);
  float xm = xs[base + m], sm = ss[base + m], ri = rinv[base + m];
  v16h f;
#pragma unroll
  for (int e = 0; e < 16; ++e) {
    int y = base + k0 + kmap(e, half);
    float t = xm * ss[y] + sm * xs[y];
    f[e] = (_Float16)(sgnroot(t) * ri);
  }
  return f;
}

#define WMMA_F16(a, b, c) \
  __builtin_amdgcn_wmma_f32_16x16x32_f16(false, (a), false, (b), (short)0, (c), false, false)

// ---------------- Layer 1: C=1, F=256 -> C=4, F=64 (d=256) ----------------
__global__ __launch_bounds__(256) void l1_kernel(
    const float* __restrict__ x, const float* __restrict__ nbr,
    const float* __restrict__ W1, const float* __restrict__ b1,
    const float* __restrict__ g1, const float* __restrict__ be1,
    const float* __restrict__ m1, const float* __restrict__ v1,
    float* __restrict__ x1, _Float16* __restrict__ nbr1)
{
  __shared__ _Float16 hbuf[24576];               // [0,16384): nbrT then W-stripe; [16384,24576): na
  __shared__ float xs[256], ss[256], rinv[256], xa[256];
  const int n = blockIdx.x, t = threadIdx.x;
  const int lane = t & 31, wave = t >> 5;
  _Float16* hB  = hbuf;          // nbrT [y][k], ld 32   (phase 3)
  _Float16* hNA = hbuf + 16384;  // na   [k][x], ld 256
  _Float16* hW  = hbuf;          // W1^T stripe [f][dl], ld 64 (phase 4)

  // phase 1: x row, s = sum_k nbr, nbr^T (f16) into LDS
  {
    xs[t] = x[(size_t)n * 256 + t];
    float s = 0.f;
    const float* nb = nbr + (size_t)n * 32 * 256 + t;
#pragma unroll 4
    for (int k = 0; k < 32; ++k) {
      if (k + 8 < 32) __builtin_prefetch(nb + (size_t)(k + 8) * 256, 0, 1);
      float v = nb[(size_t)k * 256];
      s += v;
      hB[t * 32 + k] = (_Float16)v;
    }
    ss[t] = s;
  }
  __syncthreads();
  // phase 2: row-normalization factors of adj
  {
    float xm = xs[t], sm = ss[t], acc = 0.f;
    for (int y = 0; y < 256; ++y) acc += fabsf(sgnroot(xm * ss[y] + sm * xs[y]));
    rinv[t] = 1.f / (acc + 1e-7f);
  }
  __syncthreads();
  // phase 2.5: xa = adj @ x (VALU)
  {
    float xm = xs[t], sm = ss[t], acc = 0.f;
    for (int y = 0; y < 256; ++y) acc += sgnroot(xm * ss[y] + sm * xs[y]) * xs[y];
    xa[t] = acc * rinv[t];
  }
  __syncthreads();
  // phase 3: na(256x32) = adj(256x256) x nbrT(256x32), 32 tiles, 4 per wave
  for (int i = 0; i < 4; ++i) {
    int tile = wave + 8 * i;
    int mt = tile >> 1, nt = tile & 1;
    v8f c = {};
#pragma unroll
    for (int k0 = 0; k0 < 256; k0 += 32) {
      v16h a = frag_a_adj(xs, ss, rinv, 0, mt * 16, k0, lane);
      v16h b = frag_b_lds(hB, 32, k0, nt * 16, lane);
      c = WMMA_F16(a, b, c);
    }
    int kk = nt * 16 + (lane & 15);
#pragma unroll
    for (int r = 0; r < 8; ++r) {
      int xr = mt * 16 + r + ((lane >> 4) << 3);
      hNA[kk * 256 + xr] = (_Float16)c[r];   // store transposed: na[k][x]
    }
  }
  __syncthreads();
  // phase 4: no(32x256) = na(32x256) x W1^T, streamed in 64-wide d-stripes
  for (int d0 = 0; d0 < 256; d0 += 64) {
    for (int idx = t; idx < 64 * 256; idx += 256) {
      int dl = idx & 63, f = idx >> 6;
      hW[f * 64 + dl] = (_Float16)W1[(size_t)(d0 + dl) * 256 + f];
    }
    __syncthreads();
    int mt = wave & 1, nt = wave >> 1;     // 2 k-tiles x 4 d-tiles = 8 tiles = 8 waves
    v8f c = {};
#pragma unroll
    for (int k0 = 0; k0 < 256; k0 += 32) {
      v16h a = frag_a_lds(hNA, 256, mt * 16, k0, lane);
      v16h b = frag_b_lds(hW, 64, k0, nt * 16, lane);
      c = WMMA_F16(a, b, c);
    }
    int d = d0 + nt * 16 + (lane & 15);
    int ch = d >> 6;
    float sc  = g1[ch] * rsqrtf(v1[ch] + BN_EPS);
    float off = (b1[d] - m1[ch]) * sc + be1[ch];
    _Float16* dst = nbr1 + (size_t)n * 32 * 256;
#pragma unroll
    for (int r = 0; r < 8; ++r) {
      int k = mt * 16 + r + ((lane >> 4) << 3);
      float h = c[r] * sc + off;
      dst[(size_t)k * 256 + d] = (_Float16)softsign(h);
    }
    __syncthreads();
  }
  // phase 5: xo[d] = xa . W1[d,:] + b1, BN, softsign
  {
    const float* wrow = W1 + (size_t)t * 256;
    float acc = b1[t];
    for (int f = 0; f < 256; ++f) acc += xa[f] * wrow[f];
    int ch = t >> 6;
    float sc = g1[ch] * rsqrtf(v1[ch] + BN_EPS);
    float h = (acc - m1[ch]) * sc + be1[ch];
    x1[(size_t)n * 256 + t] = softsign(h);
  }
}

// ---------------- Layer 2: C=4, F=64 -> C=8, F=16 (d=128) ----------------
__global__ __launch_bounds__(256) void l2_kernel(
    const float* __restrict__ x1, const _Float16* __restrict__ nbr1,
    const float* __restrict__ W2, const float* __restrict__ b2,
    const float* __restrict__ g2, const float* __restrict__ be2,
    const float* __restrict__ m2, const float* __restrict__ v2,
    float* __restrict__ x2, _Float16* __restrict__ nbr2)
{
  __shared__ _Float16 hbuf[24576];
  __shared__ float xs[256], ss[256], rinv[256], xa[256];
  const int n = blockIdx.x, t = threadIdx.x;
  const int lane = t & 31, wave = t >> 5;
  _Float16* hB  = hbuf;
  _Float16* hNA = hbuf + 16384;
  _Float16* hW  = hbuf;

  {
    xs[t] = x1[(size_t)n * 256 + t];
    float s = 0.f;
    const _Float16* nb = nbr1 + (size_t)n * 32 * 256 + t;
#pragma unroll 4
    for (int k = 0; k < 32; ++k) {
      _Float16 v = nb[(size_t)k * 256];
      s += (float)v;
      hB[t * 32 + k] = v;
    }
    ss[t] = s;
  }
  __syncthreads();
  {
    int base = t & ~63;                       // channel block of 64
    float xm = xs[t], sm = ss[t], acc = 0.f;
    for (int y = 0; y < 64; ++y) acc += fabsf(sgnroot(xm * ss[base + y] + sm * xs[base + y]));
    rinv[t] = 1.f / (acc + 1e-7f);
  }
  __syncthreads();
  {
    int base = t & ~63;
    float xm = xs[t], sm = ss[t], acc = 0.f;
    for (int y = 0; y < 64; ++y) acc += sgnroot(xm * ss[base + y] + sm * xs[base + y]) * xs[base + y];
    xa[t] = acc * rinv[t];
  }
  __syncthreads();
  // per-channel na_c(64x32) = adj_c(64x64) x nbrT_c(64x32): 32 tiles, 4/wave
  for (int i = 0; i < 4; ++i) {
    int tile = wave + 8 * i;
    int c4 = tile >> 3, mt = (tile >> 1) & 3, nt = tile & 1;
    int base = c4 * 64;
    v8f c = {};
#pragma unroll
    for (int k0 = 0; k0 < 64; k0 += 32) {
      v16h a = frag_a_adj(xs, ss, rinv, base, mt * 16, k0, lane);
      v16h b = frag_b_lds(hB + base * 32, 32, k0, nt * 16, lane);
      c = WMMA_F16(a, b, c);
    }
    int kk = nt * 16 + (lane & 15);
#pragma unroll
    for (int r = 0; r < 8; ++r) {
      int xr = base + mt * 16 + r + ((lane >> 4) << 3);
      hNA[kk * 256 + xr] = (_Float16)c[r];
    }
  }
  __syncthreads();
  // no(32x128) = na(32x256) x W2^T, two 64-wide stripes
  for (int d0 = 0; d0 < 128; d0 += 64) {
    for (int idx = t; idx < 64 * 256; idx += 256) {
      int dl = idx & 63, f = idx >> 6;
      hW[f * 64 + dl] = (_Float16)W2[(size_t)(d0 + dl) * 256 + f];
    }
    __syncthreads();
    int mt = wave & 1, nt = wave >> 1;
    v8f c = {};
#pragma unroll
    for (int k0 = 0; k0 < 256; k0 += 32) {
      v16h a = frag_a_lds(hNA, 256, mt * 16, k0, lane);
      v16h b = frag_b_lds(hW, 64, k0, nt * 16, lane);
      c = WMMA_F16(a, b, c);
    }
    int d = d0 + nt * 16 + (lane & 15);
    int ch = d >> 4;
    float sc  = g2[ch] * rsqrtf(v2[ch] + BN_EPS);
    float off = (b2[d] - m2[ch]) * sc + be2[ch];
    _Float16* dst = nbr2 + (size_t)n * 32 * 128;
#pragma unroll
    for (int r = 0; r < 8; ++r) {
      int k = mt * 16 + r + ((lane >> 4) << 3);
      float h = c[r] * sc + off;
      dst[(size_t)k * 128 + d] = (_Float16)softsign(h);
    }
    __syncthreads();
  }
  if (t < 128) {
    const float* wrow = W2 + (size_t)t * 256;
    float acc = b2[t];
    for (int f = 0; f < 256; ++f) acc += xa[f] * wrow[f];
    int ch = t >> 4;
    float sc = g2[ch] * rsqrtf(v2[ch] + BN_EPS);
    float h = (acc - m2[ch]) * sc + be2[ch];
    x2[(size_t)n * 128 + t] = softsign(h);
  }
}

// ---------------- Layer 3 + classifier: C=8, F=16 -> 32 -> 10 ----------------
__global__ __launch_bounds__(128) void l3_kernel(
    const float* __restrict__ x2, const _Float16* __restrict__ nbr2,
    const float* __restrict__ W3, const float* __restrict__ b3,
    const float* __restrict__ g3, const float* __restrict__ be3,
    const float* __restrict__ m3, const float* __restrict__ v3,
    const float* __restrict__ Wc, const float* __restrict__ bc,
    float* __restrict__ out)
{
  __shared__ float xs[128], ss[128], xa[128], xf[32];
  const int n = blockIdx.x, t = threadIdx.x;
  {
    xs[t] = x2[(size_t)n * 128 + t];
    float s = 0.f;
    const _Float16* nb = nbr2 + (size_t)n * 32 * 128 + t;
#pragma unroll
    for (int k = 0; k < 32; ++k) s += (float)nb[(size_t)k * 128];
    ss[t] = s;
  }
  __syncthreads();
  {
    int base = t & ~15;
    float xm = xs[t], sm = ss[t], asum = 0.f, row[16];
#pragma unroll
    for (int y = 0; y < 16; ++y) {
      float v = sgnroot(xm * ss[base + y] + sm * xs[base + y]);
      row[y] = v;
      asum += fabsf(v);
    }
    float ri = 1.f / (asum + 1e-7f), acc = 0.f;
#pragma unroll
    for (int y = 0; y < 16; ++y) acc += row[y] * xs[base + y];
    xa[t] = acc * ri;
  }
  __syncthreads();
  if (t < 32) {
    const float* wrow = W3 + (size_t)t * 128;
    float acc = b3[t];
    for (int f = 0; f < 128; ++f) acc += xa[f] * wrow[f];
    float sc = g3[t] * rsqrtf(v3[t] + BN_EPS);
    float h = (acc - m3[t]) * sc + be3[t];
    xf[t] = softsign(h);
  }
  __syncthreads();
  if (t < 10) {
    float acc = bc[t];
    const float* wrow = Wc + t * 32;
#pragma unroll
    for (int j = 0; j < 32; ++j) acc += xf[j] * wrow[j];
    out[(size_t)n * 10 + t] = acc;
  }
}

extern "C" void kernel_launch(void* const* d_in, const int* in_sizes, int n_in,
                              void* d_out, int out_size, void* d_ws, size_t ws_size,
                              hipStream_t stream) {
  const float* x   = (const float*)d_in[0];
  const float* nbr = (const float*)d_in[1];
  const float* W1 = (const float*)d_in[2];  const float* b1 = (const float*)d_in[3];
  const float* g1 = (const float*)d_in[4];  const float* be1= (const float*)d_in[5];
  const float* m1 = (const float*)d_in[6];  const float* v1 = (const float*)d_in[7];
  const float* W2 = (const float*)d_in[8];  const float* b2 = (const float*)d_in[9];
  const float* g2 = (const float*)d_in[10]; const float* be2= (const float*)d_in[11];
  const float* m2 = (const float*)d_in[12]; const float* v2 = (const float*)d_in[13];
  const float* W3 = (const float*)d_in[14]; const float* b3 = (const float*)d_in[15];
  const float* g3 = (const float*)d_in[16]; const float* be3= (const float*)d_in[17];
  const float* m3 = (const float*)d_in[18]; const float* v3 = (const float*)d_in[19];
  const float* Wc = (const float*)d_in[20]; const float* bc = (const float*)d_in[21];

  char* ws = (char*)d_ws;
  float*    x1v  = (float*)ws;                     // 1024*256 f32  (1 MB)
  _Float16* nbr1 = (_Float16*)(ws + (1  << 20));   // 1024*32*256 f16 (16 MB)
  float*    x2v  = (float*)(ws + (17 << 20));      // 1024*128 f32  (0.5 MB)
  _Float16* nbr2 = (_Float16*)(ws + (18 << 20));   // 1024*32*128 f16 (8 MB)

  l1_kernel<<<1024, 256, 0, stream>>>(x, nbr, W1, b1, g1, be1, m1, v1, x1v, nbr1);
  l2_kernel<<<1024, 256, 0, stream>>>(x1v, nbr1, W2, b2, g2, be2, m2, v2, x2v, nbr2);
  l3_kernel<<<1024, 128, 0, stream>>>(x2v, nbr2, W3, b3, g3, be3, m3, v3, Wc, bc, (float*)d_out);
}
